// Latent_83751862272683
// MI455X (gfx1250) — matrix operand order repaired and checked by reading, not compiled, not run on hardware
//
#include <hip/hip_runtime.h>
#include <hip/hip_bf16.h>
#include <math.h>

typedef __bf16 bf16_t;
typedef __attribute__((ext_vector_type(16))) __bf16 v16bf;
typedef __attribute__((ext_vector_type(8)))  float  v8f;
typedef __attribute__((ext_vector_type(4)))  float  v4f;

#define N_ROWS 16384   // 16 * 32 * 32
#define M_ROWS 8192
#define CDIM   128
#define BPAD   136     // 128 + 8 bf16 pad => 272B row stride (bank-rotating)

#if __has_builtin(__builtin_amdgcn_exp2f)
#define EXP2F(x) __builtin_amdgcn_exp2f(x)
#else
#define EXP2F(x) exp2f(x)
#endif

union FragU { v16bf v; v4f f[2]; };

// A fragment (16x32 bf16): lane l holds M = l&15; half g = l>>4;
// VGPR0-3 = K g*8+0..7 (contig 16B), VGPR4-7 = K 16+g*8+0..7 (contig 16B)
__device__ __forceinline__ v16bf load_frag_A(const bf16_t* __restrict__ P, int row0,
                                             int kk, int lane) {
  const int row = row0 + (lane & 15);
  const int g   = lane >> 4;
  const bf16_t* p = P + (size_t)row * CDIM + kk * 32 + g * 8;
  FragU u;
  u.f[0] = *reinterpret_cast<const v4f*>(p);
  u.f[1] = *reinterpret_cast<const v4f*>(p + 16);
  return u.v;
}

// B fragment (32x16 bf16) read from the LDS-staged e tile:
// lane l holds N = l&15; half g = l>>4; VGPR0-3 = K g*8+0..7, VGPR4-7 = K 16+g*8+0..7
__device__ __forceinline__ v16bf load_frag_B_lds(const bf16_t* sB, int rowloc,
                                                 int kk, int lane) {
  const int row = rowloc + (lane & 15);
  const int g   = lane >> 4;
  const bf16_t* p = sB + row * BPAD + kk * 32 + g * 8;
  FragU u;
  u.f[0] = *reinterpret_cast<const v4f*>(p);
  u.f[1] = *reinterpret_cast<const v4f*>(p + 16);
  return u.v;
}

// ---- prep: bf16 copies + squared norms -------------------------------------
__global__ void prep_z(const float* __restrict__ z, bf16_t* __restrict__ zf,
                       float* __restrict__ zsq) {
  __shared__ float red[128];
  const int n = blockIdx.x;         // flattened (b, h, w)
  const int c = threadIdx.x;        // channel
  const int b  = n >> 10;
  const int hw = n & 1023;
  const float v = z[(size_t)b * 131072 + (size_t)c * 1024 + hw];
  zf[(size_t)n * CDIM + c] = (bf16_t)v;
  red[c] = v * v;
  __syncthreads();
  for (int s = 64; s > 0; s >>= 1) { if (c < s) red[c] += red[c + s]; __syncthreads(); }
  if (c == 0) zsq[n] = red[0];
}

__global__ void prep_e(const float* __restrict__ e, bf16_t* __restrict__ ef,
                       float* __restrict__ esq) {
  __shared__ float red[128];
  const int m = blockIdx.x;
  const int c = threadIdx.x;
  const float v = e[(size_t)m * CDIM + c];
  ef[(size_t)m * CDIM + c] = (bf16_t)v;
  red[c] = v * v;
  __syncthreads();
  for (int s = 64; s > 0; s >>= 1) { if (c < s) red[c] += red[c + s]; __syncthreads(); }
  if (c == 0) esq[m] = red[0];
}

// ---- init stats -------------------------------------------------------------
__global__ void init_stats(unsigned* __restrict__ rnm, unsigned* __restrict__ cnm,
                           float* __restrict__ rsum, float* __restrict__ csum) {
  const int i = blockIdx.x * 256 + threadIdx.x;
  if (i < N_ROWS) { rnm[i] = 0x7F800000u; rsum[i] = 0.f; }   // +inf bits (min of -A)
  if (i < M_ROWS) { cnm[i] = 0x7F800000u; csum[i] = 0.f; }
}

// ---- main tiled WMMA kernel: MODE 0 = max pass, MODE 1 = sum-exp pass -------
// All per-element score math is in the base-2 domain (v_exp_f32 is base-2).
template<int MODE>
__global__ __launch_bounds__(256)
void tile_kernel(const bf16_t* __restrict__ zf, const bf16_t* __restrict__ ef,
                 const float* __restrict__ zsq, const float* __restrict__ esq,
                 unsigned* __restrict__ rnm, unsigned* __restrict__ cnm,
                 float* __restrict__ rsum, float* __restrict__ csum,
                 const float* __restrict__ log_sigma) {
  __shared__ bf16_t   sB[128 * BPAD];   // staged e tile, padded rows
  __shared__ unsigned scol_u[128];
  __shared__ float    scol_f[128];
  const int tid  = threadIdx.x;
  const int lane = tid & 31;
  const int wv   = tid >> 5;         // 8 waves, each owns 16 rows
  const int lq   = lane & 15;
  const int half = lane >> 4;
  const int n0 = blockIdx.x * 128;          // tile column base (codebook index)
  const int m0 = blockIdx.y * 128 + wv * 16; // this wave's row base (latent index)

  if (tid < 128) { if (MODE == 0) scol_u[tid] = 0x7F800000u; else scol_f[tid] = 0.f; }

  // ---- async-stage the 128x128 bf16 e tile into LDS (32 KB unique, once per
  // block instead of once per wave). Each thread issues 8x 16B async copies.
  {
    const int colg = tid & 15;        // 16B chunk within a 256B row
    const int rowb = tid >> 4;        // 0..15
    #pragma unroll
    for (int it = 0; it < 8; ++it) {
      const int row = it * 16 + rowb;
      const bf16_t* src = ef + (size_t)(n0 + row) * CDIM + colg * 8;
      const bf16_t* dst = &sB[row * BPAD + colg * 8];
      const unsigned           ldsoff = (unsigned)(size_t)dst;
      const unsigned long long ga     = (unsigned long long)(size_t)src;
      asm volatile("global_load_async_to_lds_b128 %0, %1, off"
                   :: "v"(ldsoff), "v"(ga) : "memory");
    }
    asm volatile("s_wait_asynccnt 0x0" ::: "memory");
  }
  __syncthreads();

  const float ls    = log_sigma[0];
  // alpha2 = log2(e) * (-1 / (2*exp(ls)^2)); T=1
  const float alpha = -0.5f * __expf(-2.f * ls) * 1.4426950408889634f;

  v8f acc[8];
  #pragma unroll
  for (int jb = 0; jb < 8; ++jb)
    #pragma unroll
    for (int i = 0; i < 8; ++i) acc[jb][i] = 0.f;

  #pragma unroll
  for (int kk = 0; kk < 4; ++kk) {          // K = 128 in 4 steps of 32
    v16bf a = load_frag_A(zf, m0, kk, lane);   // VMEM path (wave-exclusive rows)
    #pragma unroll
    for (int jb = 0; jb < 8; ++jb) {
      v16bf b = load_frag_B_lds(sB, jb * 16, kk, lane);  // LDS path (shared)
      acc[jb] = __builtin_amdgcn_wmma_f32_16x16x32_bf16(
          false, a, false, b, (short)0, acc[jb], false, false);
    }
  }

  // epilogue: D[r] holds S for (row m0 + half*8 + r, col n0 + jb*16 + lq)
  float zs[8], rmv[8], racc[8];
  #pragma unroll
  for (int r = 0; r < 8; ++r) {
    const int m = m0 + half * 8 + r;
    zs[r]   = zsq[m];
    racc[r] = (MODE == 0) ? -INFINITY : 0.f;
    if (MODE == 1) rmv[r] = -__uint_as_float(rnm[m]);
  }

  #pragma unroll
  for (int jb = 0; jb < 8; ++jb) {
    const int n = n0 + jb * 16 + lq;
    const float esqv = esq[n];
    const float cmv  = (MODE == 1) ? -__uint_as_float(cnm[n]) : 0.f;
    float cacc = (MODE == 0) ? -INFINITY : 0.f;
    #pragma unroll
    for (int r = 0; r < 8; ++r) {
      float d  = zs[r] + esqv - 2.f * acc[jb][r];
      d        = fmaxf(d, 0.f);
      float Av = alpha * d;                 // base-2 score, Av <= 0 always
      if (MODE == 0) { racc[r] = fmaxf(racc[r], Av); cacc = fmaxf(cacc, Av); }
      else           { racc[r] += EXP2F(Av - rmv[r]); cacc += EXP2F(Av - cmv); }
    }
    // fold the two lane-halves (same column, other 8 rows)
    float o = __shfl_xor(cacc, 16, 32);
    cacc = (MODE == 0) ? fmaxf(cacc, o) : (cacc + o);
    if (half == 0) {
      if (MODE == 0) atomicMin(&scol_u[jb * 16 + lq], __float_as_uint(-cacc));
      else           atomicAdd(&scol_f[jb * 16 + lq], cacc);
    }
  }

  // reduce per-row stats across the 16 lanes of each half
  #pragma unroll
  for (int r = 0; r < 8; ++r) {
    #pragma unroll
    for (int off = 8; off >= 1; off >>= 1) {
      float o = __shfl_xor(racc[r], off, 32);
      racc[r] = (MODE == 0) ? fmaxf(racc[r], o) : (racc[r] + o);
    }
  }
  if (lq == 0) {
    #pragma unroll
    for (int r = 0; r < 8; ++r) {
      const int m = m0 + half * 8 + r;
      if (MODE == 0) atomicMin(&rnm[m], __float_as_uint(-racc[r]));
      else           atomicAdd(&rsum[m], racc[r]);
    }
  }

  __syncthreads();
  if (tid < 128) {
    const int n = n0 + tid;
    if (MODE == 0) atomicMin(&cnm[n], scol_u[tid]);
    else           atomicAdd(&csum[n], scol_f[tid]);
  }
}

// ---- finalize (convert base-2 maxes back with one ln2 multiply) -------------
__global__ __launch_bounds__(256)
void finalize(const unsigned* __restrict__ rnm, const float* __restrict__ rsum,
              const unsigned* __restrict__ cnm, const float* __restrict__ csum,
              const float* __restrict__ log_sigma, float* __restrict__ out) {
  __shared__ double red[256];
  const int t = threadIdx.x;
  const float LN2 = 0.6931471805599453f;
  double sN = 0.0;
  for (int n = t; n < N_ROWS; n += 256)
    sN += (double)(-__uint_as_float(rnm[n]) * LN2 + logf(rsum[n]));
  red[t] = sN; __syncthreads();
  for (int s = 128; s > 0; s >>= 1) { if (t < s) red[t] += red[t + s]; __syncthreads(); }
  const double sumN = red[0];
  __syncthreads();
  double sM = 0.0;
  for (int m = t; m < M_ROWS; m += 256)
    sM += (double)(-__uint_as_float(cnm[m]) * LN2 + logf(csum[m]));
  red[t] = sM; __syncthreads();
  for (int s = 128; s > 0; s >>= 1) { if (t < s) red[t] += red[t + s]; __syncthreads(); }
  if (t == 0) {
    const float ls  = log_sigma[0];
    const float cst = 0.5f * (float)CDIM * (2.f * ls - 1.f);
    out[0] = (float)(-(sumN / (double)N_ROWS)) + cst + logf((float)M_ROWS);
    out[1] = (float)(-(red[0] / (double)M_ROWS)) + cst + logf((float)N_ROWS);
  }
}

extern "C" void kernel_launch(void* const* d_in, const int* in_sizes, int n_in,
                              void* d_out, int out_size, void* d_ws, size_t ws_size,
                              hipStream_t stream) {
  const float* z  = (const float*)d_in[0];   // [16,128,32,32]
  const float* e  = (const float*)d_in[1];   // [8192,128]
  const float* lg = (const float*)d_in[2];   // [1]
  float* out = (float*)d_out;

  char* w = (char*)d_ws;
  bf16_t*   zf   = (bf16_t*)(w);                       // 4,194,304 B
  bf16_t*   ef   = (bf16_t*)(w + 4194304);             // 2,097,152 B
  float*    zsq  = (float*)(w + 6291456);              //    65,536 B
  float*    esq  = (float*)(w + 6356992);              //    32,768 B
  unsigned* rnm  = (unsigned*)(w + 6389760);           //    65,536 B
  unsigned* cnm  = (unsigned*)(w + 6455296);           //    32,768 B
  float*    rsum = (float*)(w + 6488064);              //    65,536 B
  float*    csum = (float*)(w + 6553600);              //    32,768 B

  init_stats<<<(N_ROWS + 255) / 256, 256, 0, stream>>>(rnm, cnm, rsum, csum);
  prep_z<<<N_ROWS, 128, 0, stream>>>(z, zf, zsq);
  prep_e<<<M_ROWS, 128, 0, stream>>>(e, ef, esq);

  dim3 grid(M_ROWS / 128, N_ROWS / 128);   // (64, 128) tiles of 128x128
  tile_kernel<0><<<grid, 256, 0, stream>>>(zf, ef, zsq, esq, rnm, cnm, rsum, csum, lg);
  tile_kernel<1><<<grid, 256, 0, stream>>>(zf, ef, zsq, esq, rnm, cnm, rsum, csum, lg);

  finalize<<<1, 256, 0, stream>>>(rnm, rsum, cnm, csum, lg, out);
}
